// HGATSolver_76270029242973
// MI455X (gfx1250) — matrix operand order repaired and checked by reading, not compiled, not run on hardware
//
#include <hip/hip_runtime.h>
#include <math.h>

typedef __attribute__((ext_vector_type(16))) __bf16 bvec16;
typedef __attribute__((ext_vector_type(8)))  float  fvec8;

#define kNF 30000
#define kNS 10000
#define kEF 240000
#define kES 80000
#define kEI 40000
#define kL  6
#define kHD 128

union BF16Frag { bvec16 v; uint4 q[2]; };

// ---------------------------------------------------------------------------
// Weight convert: Wt[l][n][k] = (bf16) W[l][k][n]   (transpose for WMMA B path)
// ---------------------------------------------------------------------------
__global__ void wconv_kernel(const float* __restrict__ W, __bf16* __restrict__ Wt, int total)
{
    int idx = blockIdx.x * 256 + threadIdx.x;
    if (idx >= total) return;
    int l  = idx >> 14;
    int nk = idx & 16383;
    int n = nk >> 7, k = nk & 127;
    Wt[idx] = (__bf16)W[(l << 14) + (k << 7) + n];
}

// ---------------------------------------------------------------------------
// Encoder: h0 = concat([x, pos, time_enc(dt[batch])]) @ W + b ; also copy to h
// ---------------------------------------------------------------------------
__global__ __launch_bounds__(128)
void encode_kernel(const float* __restrict__ x, int xdim,
                   const float* __restrict__ pos,
                   const float* __restrict__ dt, const int* __restrict__ batch,
                   const float* __restrict__ W, const float* __restrict__ b,
                   float* __restrict__ h0, float* __restrict__ h)
{
    __shared__ float fin[40];
    int n = blockIdx.x;
    int c = threadIdx.x;
    int indim = xdim + 2 + 32;
    if (c < xdim) {
        fin[c] = x[n * xdim + c];
    } else if (c < xdim + 2) {
        fin[c] = pos[n * 2 + (c - xdim)];
    } else if (c < indim) {
        int t = c - xdim - 2;
        int j = t >> 1;
        float coef = -logf(10000.0f) / (16.0f - 1e-6f);
        float s = dt[batch[n]] * expf((float)j * coef);
        fin[c] = (t & 1) ? cosf(s) : sinf(s);
    }
    __syncthreads();
    float acc = b[c];
    for (int i = 0; i < indim; ++i) acc += fin[i] * W[i * 128 + c];
    h0[(size_t)n * 128 + c] = acc;
    h [(size_t)n * 128 + c] = acc;
}

// ---------------------------------------------------------------------------
// LayerNorm -> bf16 output (feeds WMMA GEMMs)
// ---------------------------------------------------------------------------
__global__ __launch_bounds__(128)
void ln_kernel(const float* __restrict__ h, const float* __restrict__ g,
               const float* __restrict__ b, __bf16* __restrict__ out)
{
    __shared__ float red[128];
    int n = blockIdx.x, c = threadIdx.x;
    float x = h[(size_t)n * 128 + c];
    red[c] = x; __syncthreads();
    for (int s = 64; s > 0; s >>= 1) { if (c < s) red[c] += red[c + s]; __syncthreads(); }
    float mu = red[0] * (1.0f / 128.0f);
    __syncthreads();
    float d = x - mu;
    red[c] = d * d; __syncthreads();
    for (int s = 64; s > 0; s >>= 1) { if (c < s) red[c] += red[c + s]; __syncthreads(); }
    float var = red[0] * (1.0f / 128.0f);
    float y = d * rsqrtf(var + 1e-5f) * g[c] + b[c];
    out[(size_t)n * 128 + c] = (__bf16)y;
}

// ---------------------------------------------------------------------------
// WMMA GEMM: C[M,128] = A[M,128] @ W[128,128] + bias ; A bf16, Wt = W^T bf16.
// Block = 4 waves; wave owns cols [32w,32w+32) as two 16x16 tiles, K=128 via
// 4x v_wmma_f32_16x16x32_bf16 per tile.  B fragments live in registers for
// the whole block; A fragments are double-buffered so the next row-tile's 8
// b128 loads overlap the current tile's 8 WMMAs.
// ---------------------------------------------------------------------------
__device__ __forceinline__ void loadAFrag(BF16Frag (&a)[4], const __bf16* __restrict__ A,
                                          int tile, int n, int kh)
{
    const __bf16* arow = A + ((size_t)tile * 16 + n) * 128;
#pragma unroll
    for (int kc = 0; kc < 4; ++kc) {
        a[kc].q[0] = *(const uint4*)(arow + kc * 32 + 8 * kh);
        a[kc].q[1] = *(const uint4*)(arow + kc * 32 + 16 + 8 * kh);
    }
}

__device__ __forceinline__ void tileCompute(const BF16Frag (&a)[4], const BF16Frag (&b)[2][4],
                                            float bias0, float bias1, __bf16* __restrict__ C,
                                            int tile, int col0, int n, int kh)
{
    fvec8 c0 = {}, c1 = {};
#pragma unroll
    for (int kc = 0; kc < 4; ++kc) {
        c0 = __builtin_amdgcn_wmma_f32_16x16x32_bf16(false, a[kc].v, false, b[0][kc].v,
                                                     (short)0, c0, false, false);
        c1 = __builtin_amdgcn_wmma_f32_16x16x32_bf16(false, a[kc].v, false, b[1][kc].v,
                                                     (short)0, c1, false, false);
    }
    // C/D layout: lane -> col (lane&15), rows 8*kh + v
    __bf16* crow = C + ((size_t)tile * 16 + 8 * kh) * 128;
#pragma unroll
    for (int v = 0; v < 8; ++v) {
        crow[(size_t)v * 128 + col0 + n]      = (__bf16)(c0[v] + bias0);
        crow[(size_t)v * 128 + col0 + 16 + n] = (__bf16)(c1[v] + bias1);
    }
}

__global__ __launch_bounds__(128)
void wmma_gemm_kernel(const __bf16* __restrict__ A, const __bf16* __restrict__ Wt,
                      const float* __restrict__ bias, __bf16* __restrict__ C,
                      int Mtiles, int tilesPerBlock)
{
    const int wave = threadIdx.x >> 5;
    const int lane = threadIdx.x & 31;
    const int n    = lane & 15;
    const int kh   = lane >> 4;
    const int col0 = wave * 32;

    // B layout (16-bit, 32x16): lanes 0-15 hold K 0..7,16..23, lanes 16-31 K 8..15,24..31,
    // column = lane&15.  With Wt[n][k] this is 2 16B loads / chunk.
    BF16Frag b[2][4];
#pragma unroll
    for (int t = 0; t < 2; ++t) {
        const __bf16* wpt = Wt + (size_t)(col0 + t * 16 + n) * 128 + 8 * kh;
#pragma unroll
        for (int kc = 0; kc < 4; ++kc) {
            b[t][kc].q[0] = *(const uint4*)(wpt + kc * 32);
            b[t][kc].q[1] = *(const uint4*)(wpt + kc * 32 + 16);
        }
    }
    const float bias0 = bias[col0 + n];
    const float bias1 = bias[col0 + 16 + n];

    const int tile0 = blockIdx.x * tilesPerBlock;
    int tEnd = tile0 + tilesPerBlock;
    if (tEnd > Mtiles) tEnd = Mtiles;
    if (tile0 >= Mtiles) return;

    BF16Frag a0[4], a1[4];
    loadAFrag(a0, A, tile0, n, kh);
    int tile = tile0;
    for (;;) {
        if (tile + 1 < tEnd) loadAFrag(a1, A, tile + 1, n, kh);
        tileCompute(a0, b, bias0, bias1, C, tile, col0, n, kh);
        if (++tile >= tEnd) break;
        if (tile + 1 < tEnd) loadAFrag(a0, A, tile + 1, n, kh);
        tileCompute(a1, b, bias0, bias1, C, tile, col0, n, kh);
        if (++tile >= tEnd) break;
    }
}

// ---------------------------------------------------------------------------
// Softmax helpers
// ---------------------------------------------------------------------------
__device__ inline void atomicMaxF(float* addr, float val)
{
    if (val >= 0.0f) atomicMax((int*)addr, __float_as_int(val));
    else             atomicMin((unsigned int*)addr, __float_as_uint(val));
}

__global__ void initsm_kernel(float* __restrict__ mx, float* __restrict__ den, int n)
{
    int i = blockIdx.x * 256 + threadIdx.x;
    if (i < n) { mx[i] = -INFINITY; den[i] = 0.0f; }
}

__global__ void initmsg_kernel(float* __restrict__ m,
                               const float* __restrict__ b0, const float* __restrict__ b1,
                               const float* __restrict__ aggw, int i0, int i1, int total)
{
    int i = blockIdx.x * 256 + threadIdx.x;
    if (i >= total) return;
    int c = i & 127;
    m[i] = aggw[i0] * b0[c] + aggw[i1] * b1[c];
}

// ---------------------------------------------------------------------------
// Edge pass A: score[e,h] = sum_d att[h,d]*lrelu(xl[src]+xr[dst]); segment max
// one wave per edge; edge id >= E -> virtual self-loop edge (src=dst=e-E)
// ---------------------------------------------------------------------------
__global__ __launch_bounds__(256)
void edge_score_kernel(const __bf16* __restrict__ xl, const __bf16* __restrict__ xr,
                       const int* __restrict__ srcI, const int* __restrict__ dstI,
                       int E, int Nloop, const float* __restrict__ att,
                       float* __restrict__ score, float* __restrict__ mx)
{
    int e = blockIdx.x * 8 + (threadIdx.x >> 5);
    int lane = threadIdx.x & 31;
    if (e >= E + Nloop) return;
    int s, d;
    if (e < E) { s = srcI[e]; d = dstI[e]; } else { s = d = e - E; }
    const __bf16* pl = xl + (size_t)s * 128;
    const __bf16* pr = xr + (size_t)d * 128;
    float part[4];
#pragma unroll
    for (int h = 0; h < 4; ++h) {
        int c = h * 32 + lane;
        float v = (float)pl[c] + (float)pr[c];
        v = v > 0.0f ? v : 0.2f * v;
        part[h] = v * att[c];
    }
#pragma unroll
    for (int off = 16; off; off >>= 1)
#pragma unroll
        for (int h = 0; h < 4; ++h)
            part[h] += __shfl_xor(part[h], off, 32);
    if (lane == 0) {
#pragma unroll
        for (int h = 0; h < 4; ++h) {
            score[(size_t)e * 4 + h] = part[h];
            atomicMaxF(&mx[d * 4 + h], part[h]);
        }
    }
}

// Edge pass B: ex = exp(score - max[dst]); denom += ex
__global__ void edge_exp_kernel(const int* __restrict__ dstI, int E, int Nloop,
                                float* __restrict__ score,
                                const float* __restrict__ mx, float* __restrict__ den)
{
    int i = blockIdx.x * 256 + threadIdx.x;
    if (i >= (E + Nloop) * 4) return;
    int e = i >> 2, h = i & 3;
    int d = (e < E) ? dstI[e] : (e - E);
    float ex = expf(score[i] - mx[d * 4 + h]);
    score[i] = ex;
    atomicAdd(&den[d * 4 + h], ex);
}

// Edge pass C: m[dst] += w * (ex/den) * xl[src]
__global__ __launch_bounds__(256)
void edge_scatter_kernel(const __bf16* __restrict__ xl,
                         const int* __restrict__ srcI, const int* __restrict__ dstI,
                         int E, int Nloop, const float* __restrict__ score,
                         const float* __restrict__ den, float* __restrict__ m,
                         const float* __restrict__ aggw, int aggidx)
{
    int e = blockIdx.x * 8 + (threadIdx.x >> 5);
    int lane = threadIdx.x & 31;
    if (e >= E + Nloop) return;
    float w = aggw[aggidx];
    int s, d;
    if (e < E) { s = srcI[e]; d = dstI[e]; } else { s = d = e - E; }
    const __bf16* pl = xl + (size_t)s * 128;
    float exv = 0.0f, dnv = 1.0f;
    if (lane < 4) { exv = score[(size_t)e * 4 + lane]; dnv = den[d * 4 + lane]; }
#pragma unroll
    for (int h = 0; h < 4; ++h) {
        float alpha = w * __shfl(exv, h, 32) / __shfl(dnv, h, 32);
        int c = h * 32 + lane;
        atomicAdd(&m[(size_t)d * 128 + c], alpha * (float)pl[c]);
    }
}

// Residual: h += relu(m)
__global__ void residual_kernel(float* __restrict__ h, const float* __restrict__ m, int total)
{
    int i = blockIdx.x * 256 + threadIdx.x;
    if (i < total) h[i] += fmaxf(m[i], 0.0f);
}

// ---------------------------------------------------------------------------
// Gate + decode: g = sigmoid([h0,h,phys]@gW+gb); hfin=(1-g)h0+g h; out=hfin@dW+db
// ---------------------------------------------------------------------------
__global__ __launch_bounds__(128)
void gate_dec_kernel(const float* __restrict__ h0, const float* __restrict__ h,
                     const float* __restrict__ phys,
                     const float* __restrict__ gW, const float* __restrict__ gb,
                     const float* __restrict__ dW, const float* __restrict__ db,
                     float* __restrict__ out, int odim)
{
    __shared__ float s0[128], s1[128], ph[4], hfin[128];
    int n = blockIdx.x, c = threadIdx.x;
    float h0v = h0[(size_t)n * 128 + c];
    float hv  = h [(size_t)n * 128 + c];
    s0[c] = h0v; s1[c] = hv;
    if (c < 4) ph[c] = phys[n * 4 + c];
    __syncthreads();
    float acc = gb[c];
    for (int i = 0; i < 128; ++i) acc += s0[i] * gW[i * 128 + c];
    for (int i = 0; i < 128; ++i) acc += s1[i] * gW[(128 + i) * 128 + c];
    for (int i = 0; i < 4;   ++i) acc += ph[i] * gW[(256 + i) * 128 + c];
    float g = 1.0f / (1.0f + expf(-acc));
    hfin[c] = (1.0f - g) * h0v + g * hv;
    __syncthreads();
    if (c < odim) {
        float o = db[c];
        for (int i = 0; i < 128; ++i) o += hfin[i] * dW[i * odim + c];
        out[(size_t)n * odim + c] = o;
    }
}

// ---------------------------------------------------------------------------
extern "C" void kernel_launch(void* const* d_in, const int* in_sizes, int n_in,
                              void* d_out, int out_size, void* d_ws, size_t ws_size,
                              hipStream_t stream)
{
    const float* x_f    = (const float*)d_in[0];
    const float* pos_f  = (const float*)d_in[1];
    const float* phys_f = (const float*)d_in[2];
    const float* x_s    = (const float*)d_in[3];
    const float* pos_s  = (const float*)d_in[4];
    const float* phys_s = (const float*)d_in[5];
    const float* dt     = (const float*)d_in[6];
    const int* batch_f  = (const int*)d_in[7];
    const int* batch_s  = (const int*)d_in[8];
    const int* ei_f2f   = (const int*)d_in[9];
    const int* ei_s2s   = (const int*)d_in[10];
    const int* ei_i2f   = (const int*)d_in[11];
    const int* ei_f2s   = (const int*)d_in[12];
    const float* enc_f_W = (const float*)d_in[37];
    const float* enc_f_b = (const float*)d_in[38];
    const float* enc_s_W = (const float*)d_in[39];
    const float* enc_s_b = (const float*)d_in[40];
    const float* lnf_g = (const float*)d_in[41];
    const float* lnf_b = (const float*)d_in[42];
    const float* lns_g = (const float*)d_in[43];
    const float* lns_b = (const float*)d_in[44];
    const float* agg_w = (const float*)d_in[45];
    const float* gate_f_W = (const float*)d_in[46];
    const float* gate_f_b = (const float*)d_in[47];
    const float* gate_s_W = (const float*)d_in[48];
    const float* gate_s_b = (const float*)d_in[49];
    const float* dec_f_W = (const float*)d_in[50];
    const float* dec_f_b = (const float*)d_in[51];
    const float* dec_s_W = (const float*)d_in[52];
    const float* dec_s_b = (const float*)d_in[53];

    // workspace (everything fits in 192MB L2; ~94MB used)
    char* wp = (char*)d_ws;
    auto alloc = [&](size_t bytes) -> void* {
        void* p = (void*)wp; wp += (bytes + 255) & ~(size_t)255; return p;
    };
    float*  h0f = (float*)alloc(sizeof(float) * (size_t)kNF * kHD);
    float*  h0s = (float*)alloc(sizeof(float) * (size_t)kNS * kHD);
    float*  hf  = (float*)alloc(sizeof(float) * (size_t)kNF * kHD);
    float*  hs  = (float*)alloc(sizeof(float) * (size_t)kNS * kHD);
    __bf16* Fbf = (__bf16*)alloc(sizeof(__bf16) * (size_t)kNF * kHD);
    __bf16* Sbf = (__bf16*)alloc(sizeof(__bf16) * (size_t)kNS * kHD);
    __bf16* xl  = (__bf16*)alloc(sizeof(__bf16) * (size_t)kNF * kHD);
    __bf16* xr  = (__bf16*)alloc(sizeof(__bf16) * (size_t)kNF * kHD);
    float*  score = (float*)alloc(sizeof(float) * (size_t)(kEF + kNF) * 4);
    float*  mx  = (float*)alloc(sizeof(float) * (size_t)kNF * 4);
    float*  den = (float*)alloc(sizeof(float) * (size_t)kNF * 4);
    float*  m_f = (float*)alloc(sizeof(float) * (size_t)kNF * kHD);
    float*  m_s = (float*)alloc(sizeof(float) * (size_t)kNS * kHD);
    __bf16* WtAll = (__bf16*)alloc(sizeof(__bf16) * (size_t)8 * kL * kHD * kHD);

    struct Rel {
        const float *Wl, *bl, *Wr, *br, *att, *bias;
        const int *src, *dst;
        int E, Nloop, Nsrc, Ndst;
        const __bf16 *srcbf, *dstbf;
        float* msg;
        int aggSlot;
        __bf16 *WtL, *WtR;
    } rel[4];
    // order: f2f, s2s, i2f, f2s (input indices 13,19,25,31)
    const int base[4] = {13, 19, 25, 31};
    const int* eptr[4] = {ei_f2f, ei_s2s, ei_i2f, ei_f2s};
    const int Ecnt[4] = {kEF, kES, kEI, kEI};
    const int Nloop[4] = {kNF, kNS, 0, 0};
    const int NsrcA[4] = {kNF, kNS, kNS, kNF};
    const int NdstA[4] = {kNF, kNS, kNF, kNS};
    const int aggSlot[4] = {0, 2, 1, 3};
    for (int r = 0; r < 4; ++r) {
        rel[r].Wl   = (const float*)d_in[base[r] + 0];
        rel[r].bl   = (const float*)d_in[base[r] + 1];
        rel[r].Wr   = (const float*)d_in[base[r] + 2];
        rel[r].br   = (const float*)d_in[base[r] + 3];
        rel[r].att  = (const float*)d_in[base[r] + 4];
        rel[r].bias = (const float*)d_in[base[r] + 5];
        rel[r].src  = eptr[r];
        rel[r].dst  = eptr[r] + Ecnt[r];
        rel[r].E = Ecnt[r]; rel[r].Nloop = Nloop[r];
        rel[r].Nsrc = NsrcA[r]; rel[r].Ndst = NdstA[r];
        rel[r].srcbf = (NsrcA[r] == kNF) ? Fbf : Sbf;
        rel[r].dstbf = (NdstA[r] == kNF) ? Fbf : Sbf;
        rel[r].msg  = (NdstA[r] == kNF) ? m_f : m_s;
        rel[r].aggSlot = aggSlot[r];
        rel[r].WtL = WtAll + (size_t)(2 * r + 0) * kL * kHD * kHD;
        rel[r].WtR = WtAll + (size_t)(2 * r + 1) * kL * kHD * kHD;
    }

    const int WT = kL * kHD * kHD;
    for (int r = 0; r < 4; ++r) {
        wconv_kernel<<<(WT + 255) / 256, 256, 0, stream>>>(rel[r].Wl, rel[r].WtL, WT);
        wconv_kernel<<<(WT + 255) / 256, 256, 0, stream>>>(rel[r].Wr, rel[r].WtR, WT);
    }

    encode_kernel<<<kNF, 128, 0, stream>>>(x_f, 3, pos_f, dt, batch_f, enc_f_W, enc_f_b, h0f, hf);
    encode_kernel<<<kNS, 128, 0, stream>>>(x_s, 2, pos_s, dt, batch_s, enc_s_W, enc_s_b, h0s, hs);

    for (int i = 0; i < kL; ++i) {
        ln_kernel<<<kNF, 128, 0, stream>>>(hf, lnf_g + i * 128, lnf_b + i * 128, Fbf);
        ln_kernel<<<kNS, 128, 0, stream>>>(hs, lns_g + i * 128, lns_b + i * 128, Sbf);

        // m_f = w0*bias_f2f + w1*bias_i2f ; m_s = w2*bias_s2s + w3*bias_f2s
        initmsg_kernel<<<(kNF * 128 + 255) / 256, 256, 0, stream>>>(
            m_f, rel[0].bias + i * 128, rel[2].bias + i * 128, agg_w, i * 4 + 0, i * 4 + 1, kNF * 128);
        initmsg_kernel<<<(kNS * 128 + 255) / 256, 256, 0, stream>>>(
            m_s, rel[1].bias + i * 128, rel[3].bias + i * 128, agg_w, i * 4 + 2, i * 4 + 3, kNS * 128);

        for (int r = 0; r < 4; ++r) {
            const Rel& R = rel[r];
            int mtS = R.Nsrc / 16, mtD = R.Ndst / 16;
            wmma_gemm_kernel<<<(mtS + 7) / 8, 128, 0, stream>>>(
                R.srcbf, R.WtL + (size_t)i * 16384, R.bl + i * 128, xl, mtS, 8);
            wmma_gemm_kernel<<<(mtD + 7) / 8, 128, 0, stream>>>(
                R.dstbf, R.WtR + (size_t)i * 16384, R.br + i * 128, xr, mtD, 8);

            initsm_kernel<<<(R.Ndst * 4 + 255) / 256, 256, 0, stream>>>(mx, den, R.Ndst * 4);
            int Etot = R.E + R.Nloop;
            edge_score_kernel<<<(Etot + 7) / 8, 256, 0, stream>>>(
                xl, xr, R.src, R.dst, R.E, R.Nloop, R.att + i * 128, score, mx);
            edge_exp_kernel<<<(Etot * 4 + 255) / 256, 256, 0, stream>>>(
                R.dst, R.E, R.Nloop, score, mx, den);
            edge_scatter_kernel<<<(Etot + 7) / 8, 256, 0, stream>>>(
                xl, R.src, R.dst, R.E, R.Nloop, score, den, R.msg, agg_w, i * 4 + R.aggSlot);
        }

        residual_kernel<<<(kNF * 128 + 255) / 256, 256, 0, stream>>>(hf, m_f, kNF * 128);
        residual_kernel<<<(kNS * 128 + 255) / 256, 256, 0, stream>>>(hs, m_s, kNS * 128);
    }

    float* outF = (float*)d_out;
    float* outS = outF + (size_t)kNF * 3;
    gate_dec_kernel<<<kNF, 128, 0, stream>>>(h0f, hf, phys_f, gate_f_W, gate_f_b,
                                             dec_f_W, dec_f_b, outF, 3);
    gate_dec_kernel<<<kNS, 128, 0, stream>>>(h0s, hs, phys_s, gate_s_W, gate_s_b,
                                             dec_s_W, dec_s_b, outS, 2);
}
